// LSTM_Local_Attention_47261820125595
// MI455X (gfx1250) — compile-verified, matrix operand order checked by reading, not run on hardware
//
#include <hip/hip_runtime.h>
#include <hip/hip_bf16.h>
#include <cstdint>

#define Bc 16
#define Sc 256
#define Tc 64
#define Ec 256
#define Hc 512
#define DHc 1024
#define VTGT 32000

typedef __attribute__((ext_vector_type(16))) __bf16 v16bf;
typedef __attribute__((ext_vector_type(8)))  float  v8f;

__device__ __forceinline__ v8f wmma_bf16(v16bf a, v16bf b, v8f c) {
  return __builtin_amdgcn_wmma_f32_16x16x32_bf16(false, a, false, b, (short)0, c, false, false);
}
__device__ __forceinline__ float sigm(float x) { return 1.f / (1.f + __expf(-x)); }

// ---------------- elementwise helpers ----------------
__global__ void cvt_bf16(const float* __restrict__ s, __bf16* __restrict__ d, size_t n) {
  for (size_t i = (size_t)blockIdx.x * blockDim.x + threadIdx.x; i < n;
       i += (size_t)gridDim.x * blockDim.x)
    d[i] = (__bf16)s[i];
}

// xs[s,b,e] = src_emb[x[b, S-1-s], e]  (reversed source)
__global__ void gather_src(const int* __restrict__ x, const float* __restrict__ emb,
                           __bf16* __restrict__ xs) {
  int i = blockIdx.x * blockDim.x + threadIdx.x;       // S*B*E
  if (i >= Sc * Bc * Ec) return;
  int e = i % Ec, sb = i / Ec, b = sb % Bc, s = sb / Bc;
  int tok = x[b * Sc + (Sc - 1 - s)];
  xs[i] = (__bf16)emb[(size_t)tok * Ec + e];
}

// emby[t,b,e] = tgt_emb[y[b,t], e]   (teacher forcing, t < T-1)
__global__ void gather_tgt(const int* __restrict__ y, const float* __restrict__ emb,
                           __bf16* __restrict__ emby) {
  int i = blockIdx.x * blockDim.x + threadIdx.x;       // (T-1)*B*E
  if (i >= (Tc - 1) * Bc * Ec) return;
  int e = i % Ec, tb = i / Ec, b = tb % Bc, t = tb / Bc;
  int tok = y[b * Tc + t];
  emby[i] = (__bf16)emb[(size_t)tok * Ec + e];
}

// ---------------- generic WMMA GEMM:  C[M,N] = act(A[M,K] @ W[N,K]^T + bias) ----------------
// 4 waves/block share one 16x32 A tile staged in LDS; each wave owns one 16x16 N-tile.
// MODE: 0 = f32 out, 1 = bf16 out.  ACT: 0 = none, 1 = tanh.
template <int MODE, int ACT>
__global__ __launch_bounds__(128) void gemm_wmma(
    const __bf16* __restrict__ A, int lda, const __bf16* __restrict__ W, int ldw,
    float* __restrict__ Cf, __bf16* __restrict__ Cb, int ldc,
    const float* __restrict__ bias, int K) {
  __shared__ __align__(64) __bf16 As[16 * 32];
  const int wave = threadIdx.x >> 5;
  const int lane = threadIdx.x & 31;
  const int mt = blockIdx.x;
  const int nt = blockIdx.y * 4 + wave;
  const int arow = lane & 15, asel = (lane >> 4) * 16;
  const __bf16* wrow = W + (size_t)(nt * 16 + arow) * ldw + asel;
  const __bf16* ag = A + (size_t)(mt * 16) * lda;
  v8f acc = {};
  const int le = threadIdx.x * 4, lr = le >> 5, lc = le & 31;
  for (int k0 = 0; k0 < K; k0 += 32) {
    *(uint64_t*)&As[le] = *(const uint64_t*)&ag[(size_t)lr * lda + k0 + lc];
    __syncthreads();
    v16bf af = *(const v16bf*)&As[arow * 32 + asel];
    v16bf bf = *(const v16bf*)&wrow[k0];
    acc = wmma_bf16(af, bf, acc);
    __syncthreads();
  }
  const int nc = nt * 16 + (lane & 15);
  const int m0 = (lane >> 4) * 8;
  float bv = bias ? bias[nc] : 0.f;
#pragma unroll
  for (int r = 0; r < 8; ++r) {
    float v = acc[r] + bv;
    if (ACT == 1) v = tanhf(v);
    size_t off = (size_t)(mt * 16 + m0 + r) * ldc + nc;
    if (MODE == 0) Cf[off] = v;
    else           Cb[off] = (__bf16)v;
  }
}

// ---------------- encoder recurrent cell (one step, both directions) ----------------
// gates = gatePre[s] (= x@Wih^T + bih) + h_prev @ Whh^T + bhh ; LSTM cell; h -> ys (f32+bf16)
__global__ __launch_bounds__(32) void enc_cell(
    const float* __restrict__ gpF, const float* __restrict__ gpB,
    const __bf16* __restrict__ WhhF, const __bf16* __restrict__ WhhB,
    const __bf16* __restrict__ hpF, const __bf16* __restrict__ hpB,
    __bf16* __restrict__ hnF, __bf16* __restrict__ hnB,
    float* __restrict__ cF, float* __restrict__ cB,
    const float* __restrict__ bhhF, const float* __restrict__ bhhB,
    float* __restrict__ ys, __bf16* __restrict__ ysbf, int t) {
  const int dir = blockIdx.y;
  const int s = dir ? (Sc - 1 - t) : t;
  const int n = blockIdx.x;                      // hidden block 0..31
  const int lane = threadIdx.x;
  const float* gp = dir ? gpB : gpF;
  const __bf16* Whh = dir ? WhhB : WhhF;
  const __bf16* hp = dir ? hpB : hpF;
  __bf16* hn = dir ? hnB : hnF;
  float* cc = dir ? cB : cF;
  const float* bhh = dir ? bhhB : bhhF;

  const int arow = lane & 15, asel = (lane >> 4) * 16;
  v8f a0 = {}, a1 = {}, a2 = {}, a3 = {};
  const __bf16* w0 = Whh + (size_t)(0 * Hc + n * 16 + arow) * Hc + asel;
  const __bf16* w1 = Whh + (size_t)(1 * Hc + n * 16 + arow) * Hc + asel;
  const __bf16* w2 = Whh + (size_t)(2 * Hc + n * 16 + arow) * Hc + asel;
  const __bf16* w3 = Whh + (size_t)(3 * Hc + n * 16 + arow) * Hc + asel;
  for (int k0 = 0; k0 < Hc; k0 += 32) {
    v16bf a = *(const v16bf*)&hp[arow * Hc + asel + k0];
    a0 = wmma_bf16(a, *(const v16bf*)&w0[k0], a0);
    a1 = wmma_bf16(a, *(const v16bf*)&w1[k0], a1);
    a2 = wmma_bf16(a, *(const v16bf*)&w2[k0], a2);
    a3 = wmma_bf16(a, *(const v16bf*)&w3[k0], a3);
  }
  const int col = lane & 15, j = n * 16 + col, m0 = (lane >> 4) * 8;
#pragma unroll
  for (int r = 0; r < 8; ++r) {
    int b = m0 + r;
    size_t grow = (size_t)(s * Bc + b) * (4 * Hc);
    float gi = a0[r] + gp[grow + 0 * Hc + j] + bhh[0 * Hc + j];
    float gf = a1[r] + gp[grow + 1 * Hc + j] + bhh[1 * Hc + j];
    float gg = a2[r] + gp[grow + 2 * Hc + j] + bhh[2 * Hc + j];
    float go = a3[r] + gp[grow + 3 * Hc + j] + bhh[3 * Hc + j];
    float cn = sigm(gf) * cc[b * Hc + j] + sigm(gi) * tanhf(gg);
    float h = sigm(go) * tanhf(cn);
    cc[b * Hc + j] = cn;
    hn[b * Hc + j] = (__bf16)h;
    size_t yo = ((size_t)s * Bc + b) * (2 * Hc) + dir * Hc + j;
    ys[yo] = h;
    ysbf[yo] = (__bf16)h;
  }
}

// ---------------- decoder LSTM cell (x@Wih^T + h@Whh^T + biases -> cell) ----------------
__global__ __launch_bounds__(32) void dec_cell(
    const __bf16* __restrict__ x, int Kx, const __bf16* __restrict__ Wih,
    const __bf16* __restrict__ hp, const __bf16* __restrict__ Whh,
    const float* __restrict__ bih, const float* __restrict__ bhh,
    float* __restrict__ cc, __bf16* __restrict__ hn,
    float* __restrict__ hf32, __bf16* __restrict__ hbf2, int ldh2) {
  const int n = blockIdx.x;                      // hidden block 0..63
  const int lane = threadIdx.x;
  const int arow = lane & 15, asel = (lane >> 4) * 16;
  v8f a0 = {}, a1 = {}, a2 = {}, a3 = {};
  {
    const __bf16* w0 = Wih + (size_t)(0 * DHc + n * 16 + arow) * Kx + asel;
    const __bf16* w1 = Wih + (size_t)(1 * DHc + n * 16 + arow) * Kx + asel;
    const __bf16* w2 = Wih + (size_t)(2 * DHc + n * 16 + arow) * Kx + asel;
    const __bf16* w3 = Wih + (size_t)(3 * DHc + n * 16 + arow) * Kx + asel;
    for (int k0 = 0; k0 < Kx; k0 += 32) {
      v16bf a = *(const v16bf*)&x[arow * Kx + asel + k0];
      a0 = wmma_bf16(a, *(const v16bf*)&w0[k0], a0);
      a1 = wmma_bf16(a, *(const v16bf*)&w1[k0], a1);
      a2 = wmma_bf16(a, *(const v16bf*)&w2[k0], a2);
      a3 = wmma_bf16(a, *(const v16bf*)&w3[k0], a3);
    }
  }
  {
    const __bf16* w0 = Whh + (size_t)(0 * DHc + n * 16 + arow) * DHc + asel;
    const __bf16* w1 = Whh + (size_t)(1 * DHc + n * 16 + arow) * DHc + asel;
    const __bf16* w2 = Whh + (size_t)(2 * DHc + n * 16 + arow) * DHc + asel;
    const __bf16* w3 = Whh + (size_t)(3 * DHc + n * 16 + arow) * DHc + asel;
    for (int k0 = 0; k0 < DHc; k0 += 32) {
      v16bf a = *(const v16bf*)&hp[arow * DHc + asel + k0];
      a0 = wmma_bf16(a, *(const v16bf*)&w0[k0], a0);
      a1 = wmma_bf16(a, *(const v16bf*)&w1[k0], a1);
      a2 = wmma_bf16(a, *(const v16bf*)&w2[k0], a2);
      a3 = wmma_bf16(a, *(const v16bf*)&w3[k0], a3);
    }
  }
  const int col = lane & 15, j = n * 16 + col, m0 = (lane >> 4) * 8;
#pragma unroll
  for (int r = 0; r < 8; ++r) {
    int b = m0 + r;
    float gi = a0[r] + bih[0 * DHc + j] + bhh[0 * DHc + j];
    float gf = a1[r] + bih[1 * DHc + j] + bhh[1 * DHc + j];
    float gg = a2[r] + bih[2 * DHc + j] + bhh[2 * DHc + j];
    float go = a3[r] + bih[3 * DHc + j] + bhh[3 * DHc + j];
    float cn = sigm(gf) * cc[b * DHc + j] + sigm(gi) * tanhf(gg);
    float h = sigm(go) * tanhf(cn);
    cc[b * DHc + j] = cn;
    hn[b * DHc + j] = (__bf16)h;
    if (hf32) hf32[b * DHc + j] = h;
    if (hbf2) hbf2[b * ldh2 + j] = (__bf16)h;
  }
}

// ---------------- p_t = S * sigmoid( tanhproj . v_p + b_vp ) ----------------
__global__ __launch_bounds__(32) void pt_kernel(const __bf16* __restrict__ tp,
                                                const __bf16* __restrict__ vp,
                                                const float* __restrict__ bvp,
                                                float* __restrict__ p_t) {
  int b = blockIdx.x, lane = threadIdx.x;
  float s = 0.f;
  for (int k = lane; k < DHc; k += 32) s += (float)tp[b * DHc + k] * (float)vp[k];
  for (int off = 16; off; off >>= 1) s += __shfl_down(s, off);
  if (lane == 0) p_t[b] = (float)Sc * sigm(s + bvp[0]);
}

// ---------------- fused scores / softmax*gauss / renorm / context  (one block per batch) ----------------
__global__ __launch_bounds__(256) void attn_ctx(
    const float* __restrict__ h1, const float* __restrict__ encproj,
    const float* __restrict__ encout, const float* __restrict__ p_t,
    __bf16* __restrict__ cc) {
  __shared__ float red[256];
  __shared__ float wsh[256];
  int b = blockIdx.x, s = threadIdx.x;
  const float* hp = h1 + b * DHc;
  const float* pr = encproj + ((size_t)s * Bc + b) * DHc;
  float sc = 0.f;
  for (int d = 0; d < DHc; ++d) sc += hp[d] * pr[d];
  red[s] = sc; __syncthreads();
  for (int off = 128; off; off >>= 1) { if (s < off) red[s] = fmaxf(red[s], red[s + off]); __syncthreads(); }
  float m = red[0]; __syncthreads();
  float e = __expf(sc - m);
  red[s] = e; __syncthreads();
  for (int off = 128; off; off >>= 1) { if (s < off) red[s] += red[s + off]; __syncthreads(); }
  float Z = red[0]; __syncthreads();
  float d0 = (float)s - p_t[b];
  float w = (e / Z) * __expf(-(d0 * d0) * (1.f / 50.f));   // sigma = D_WIN/2 = 5
  red[s] = w; __syncthreads();
  for (int off = 128; off; off >>= 1) { if (s < off) red[s] += red[s + off]; __syncthreads(); }
  float Ws = red[0] + 1e-10f; __syncthreads();
  wsh[s] = w / Ws; __syncthreads();
  for (int d = s; d < DHc; d += 256) {
    float acc = 0.f;
    for (int ss = 0; ss < Sc; ++ss) acc += wsh[ss] * encout[((size_t)ss * Bc + b) * DHc + d];
    cc[(size_t)b * (2 * DHc) + d] = (__bf16)acc;   // ctx half of concat(ctx, h1)
  }
}

// rnn_in = concat(emb_t, att)   [16, E+DH] bf16
__global__ void build_rnn(const __bf16* __restrict__ emby, const __bf16* __restrict__ att,
                          __bf16* __restrict__ rnn, int t) {
  int i = blockIdx.x * blockDim.x + threadIdx.x;
  if (i >= Bc * (Ec + DHc)) return;
  int b = i / (Ec + DHc), c = i % (Ec + DHc);
  rnn[i] = (c < Ec) ? emby[(size_t)t * Bc * Ec + b * Ec + c] : att[b * DHc + (c - Ec)];
}

// decoder initial state from encoder finals
__global__ void dec_init(const float* __restrict__ ys0, const float* __restrict__ ys1,
                         const float* __restrict__ c0f, const float* __restrict__ c0b,
                         const float* __restrict__ c1f, const float* __restrict__ c1b,
                         __bf16* __restrict__ h0, __bf16* __restrict__ h1,
                         float* __restrict__ c0, float* __restrict__ c1,
                         __bf16* __restrict__ attbf) {
  int i = blockIdx.x * blockDim.x + threadIdx.x;
  if (i >= Bc * DHc) return;
  int b = i / DHc, j = i % DHc;
  size_t fo = ((size_t)(Sc - 1) * Bc + b) * DHc + j;   // fwd final at s=S-1
  size_t bo = (size_t)b * DHc + j;                      // bwd final at s=0
  h0[i] = (__bf16)((j < Hc) ? ys0[fo] : ys0[bo]);
  h1[i] = (__bf16)((j < Hc) ? ys1[fo] : ys1[bo]);
  c0[i] = (j < Hc) ? c0f[b * Hc + j] : c0b[b * Hc + j - Hc];
  c1[i] = (j < Hc) ? c1f[b * Hc + j] : c1b[b * Hc + j - Hc];
  attbf[i] = (__bf16)0.f;
}

// ---------------- host orchestration ----------------
extern "C" void kernel_launch(void* const* d_in, const int* in_sizes, int n_in,
                              void* d_out, int out_size, void* d_ws, size_t ws_size,
                              hipStream_t stream) {
  (void)in_sizes; (void)out_size; (void)ws_size;
  if (n_in < 37) return;
  // flattened input order: x, y, src_emb, tgt_emb,
  //   enc0.fwd{Wih,Whh,bih,bhh}, enc0.bwd, enc1.fwd, enc1.bwd,
  //   dec0{...}, dec1{...}, W_a, W_p, b_p, v_p, b_vp, W_c, b_c, W_out, b_out
  const int*   x       = (const int*)d_in[0];
  const int*   y       = (const int*)d_in[1];
  const float* src_emb = (const float*)d_in[2];
  const float* tgt_emb = (const float*)d_in[3];
  const float* W_e[4][2], *B_e[4][2];      // [layer*2+? ] unused-style
  struct LP { const float *Wih, *Whh, *bih, *bhh; };
  LP lp[6];
  for (int i = 0; i < 6; ++i) {
    lp[i].Wih = (const float*)d_in[4 + 4 * i + 0];
    lp[i].Whh = (const float*)d_in[4 + 4 * i + 1];
    lp[i].bih = (const float*)d_in[4 + 4 * i + 2];
    lp[i].bhh = (const float*)d_in[4 + 4 * i + 3];
  }
  (void)W_e; (void)B_e;
  LP e0f = lp[0], e0b = lp[1], e1f = lp[2], e1b = lp[3], dc0 = lp[4], dc1 = lp[5];
  const float* W_a  = (const float*)d_in[28];
  const float* W_p  = (const float*)d_in[29];
  const float* b_p  = (const float*)d_in[30];
  const float* v_p  = (const float*)d_in[31];
  const float* b_vp = (const float*)d_in[32];
  const float* W_c  = (const float*)d_in[33];
  const float* b_c  = (const float*)d_in[34];
  const float* W_out= (const float*)d_in[35];
  const float* b_out= (const float*)d_in[36];
  float* out = (float*)d_out;

  // ---- workspace carve ----
  char* wp = (char*)d_ws;
  auto alloc = [&](size_t bytes) -> char* {
    char* r = wp; wp += (bytes + 255) & ~(size_t)255; return r;
  };
  auto abf = [&](size_t n) { return (__bf16*)alloc(n * sizeof(__bf16)); };
  auto af32 = [&](size_t n) { return (float*)alloc(n * sizeof(float)); };

  __bf16 *bWihE[2][2], *bWhhE[2][2];
  bWihE[0][0] = abf((size_t)4 * Hc * Ec);        bWhhE[0][0] = abf((size_t)4 * Hc * Hc);
  bWihE[0][1] = abf((size_t)4 * Hc * Ec);        bWhhE[0][1] = abf((size_t)4 * Hc * Hc);
  bWihE[1][0] = abf((size_t)4 * Hc * 2 * Hc);    bWhhE[1][0] = abf((size_t)4 * Hc * Hc);
  bWihE[1][1] = abf((size_t)4 * Hc * 2 * Hc);    bWhhE[1][1] = abf((size_t)4 * Hc * Hc);
  __bf16* bWih0 = abf((size_t)4 * DHc * (Ec + DHc));
  __bf16* bWhh0 = abf((size_t)4 * DHc * DHc);
  __bf16* bWih1 = abf((size_t)4 * DHc * DHc);
  __bf16* bWhh1 = abf((size_t)4 * DHc * DHc);
  __bf16* bWa = abf((size_t)DHc * DHc);
  __bf16* bWp = abf((size_t)DHc * DHc);
  __bf16* bvp = abf(DHc);
  __bf16* bWc = abf((size_t)DHc * 2 * DHc);
  __bf16* bWo = abf((size_t)VTGT * DHc);

  __bf16* xs0   = abf((size_t)Sc * Bc * Ec);
  __bf16* emby  = abf((size_t)(Tc - 1) * Bc * Ec);
  float*  gpF   = af32((size_t)Sc * Bc * 4 * Hc);
  float*  gpB   = af32((size_t)Sc * Bc * 4 * Hc);
  float*  ys0   = af32((size_t)Sc * Bc * 2 * Hc);
  __bf16* ys0b  = abf((size_t)Sc * Bc * 2 * Hc);
  float*  ys1   = af32((size_t)Sc * Bc * 2 * Hc);
  __bf16* ys1b  = abf((size_t)Sc * Bc * 2 * Hc);
  float*  eproj = af32((size_t)Sc * Bc * DHc);
  __bf16* henc  = abf((size_t)2 * 2 * Bc * Hc);    // [pp][dir][16*Hc]
  float*  cL0   = af32((size_t)2 * Bc * Hc);       // [dir][16*Hc]
  float*  cL1   = af32((size_t)2 * Bc * Hc);
  __bf16* h0pp  = abf((size_t)2 * Bc * DHc);
  __bf16* h1pp  = abf((size_t)2 * Bc * DHc);
  float*  c0d   = af32((size_t)Bc * DHc);
  float*  c1d   = af32((size_t)Bc * DHc);
  float*  h1f   = af32((size_t)Bc * DHc);
  __bf16* tpb   = abf((size_t)Bc * DHc);
  float*  p_t   = af32(Bc);
  __bf16* rnn   = abf((size_t)Bc * (Ec + DHc));
  __bf16* ccb   = abf((size_t)Bc * 2 * DHc);
  __bf16* attb  = abf((size_t)Bc * DHc);

  auto cvt = [&](const float* s, __bf16* d, size_t n) {
    cvt_bf16<<<dim3(1024), 256, 0, stream>>>(s, d, n);
  };
  // ---- weight conversions ----
  cvt(e0f.Wih, bWihE[0][0], (size_t)4 * Hc * Ec);     cvt(e0f.Whh, bWhhE[0][0], (size_t)4 * Hc * Hc);
  cvt(e0b.Wih, bWihE[0][1], (size_t)4 * Hc * Ec);     cvt(e0b.Whh, bWhhE[0][1], (size_t)4 * Hc * Hc);
  cvt(e1f.Wih, bWihE[1][0], (size_t)4 * Hc * 2 * Hc); cvt(e1f.Whh, bWhhE[1][0], (size_t)4 * Hc * Hc);
  cvt(e1b.Wih, bWihE[1][1], (size_t)4 * Hc * 2 * Hc); cvt(e1b.Whh, bWhhE[1][1], (size_t)4 * Hc * Hc);
  cvt(dc0.Wih, bWih0, (size_t)4 * DHc * (Ec + DHc));  cvt(dc0.Whh, bWhh0, (size_t)4 * DHc * DHc);
  cvt(dc1.Wih, bWih1, (size_t)4 * DHc * DHc);         cvt(dc1.Whh, bWhh1, (size_t)4 * DHc * DHc);
  cvt(W_a, bWa, (size_t)DHc * DHc);  cvt(W_p, bWp, (size_t)DHc * DHc);
  cvt(v_p, bvp, DHc);                cvt(W_c, bWc, (size_t)DHc * 2 * DHc);
  cvt(W_out, bWo, (size_t)VTGT * DHc);

  // ---- embeddings ----
  gather_src<<<dim3((Sc * Bc * Ec + 255) / 256), 256, 0, stream>>>(x, src_emb, xs0);
  gather_tgt<<<dim3(((Tc - 1) * Bc * Ec + 255) / 256), 256, 0, stream>>>(y, tgt_emb, emby);

  // ---- encoder ----
  const float* cbufs[2] = {nullptr, nullptr};
  for (int l = 0; l < 2; ++l) {
    const __bf16* xin = (l == 0) ? xs0 : ys0b;
    const int Kin = (l == 0) ? Ec : 2 * Hc;
    LP pf = (l == 0) ? e0f : e1f;
    LP pb = (l == 0) ? e0b : e1b;
    float* yso = (l == 0) ? ys0 : ys1;
    __bf16* ysob = (l == 0) ? ys0b : ys1b;
    float* cbuf = (l == 0) ? cL0 : cL1;
    cbufs[l] = cbuf;
    // input projections for all timesteps: [S*B, Kin] @ [Kin, 4H] (+bih)
    gemm_wmma<0, 0><<<dim3(Sc * Bc / 16, 4 * Hc / 64), 128, 0, stream>>>(
        xin, Kin, bWihE[l][0], Kin, gpF, nullptr, 4 * Hc, pf.bih, Kin);
    gemm_wmma<0, 0><<<dim3(Sc * Bc / 16, 4 * Hc / 64), 128, 0, stream>>>(
        xin, Kin, bWihE[l][1], Kin, gpB, nullptr, 4 * Hc, pb.bih, Kin);
    hipMemsetAsync(henc, 0, (size_t)2 * 2 * Bc * Hc * sizeof(__bf16), stream);
    hipMemsetAsync(cbuf, 0, (size_t)2 * Bc * Hc * sizeof(float), stream);
    int pp = 0;
    for (int t = 0; t < Sc; ++t) {
      __bf16* hp = henc + (size_t)pp * 2 * Bc * Hc;
      __bf16* hn = henc + (size_t)(1 - pp) * 2 * Bc * Hc;
      enc_cell<<<dim3(Hc / 16, 2), 32, 0, stream>>>(
          gpF, gpB, bWhhE[l][0], bWhhE[l][1],
          hp, hp + Bc * Hc, hn, hn + Bc * Hc,
          cbuf, cbuf + Bc * Hc, pf.bhh, pb.bhh, yso, ysob, t);
      pp ^= 1;
    }
  }
  // enc_proj = enc_out @ W_a^T
  gemm_wmma<0, 0><<<dim3(Sc * Bc / 16, DHc / 64), 128, 0, stream>>>(
      ys1b, DHc, bWa, DHc, eproj, nullptr, DHc, nullptr, DHc);

  // ---- decoder init ----
  dec_init<<<dim3(Bc * DHc / 256), 256, 0, stream>>>(
      ys0, ys1, cbufs[0], cbufs[0] + Bc * Hc, cbufs[1], cbufs[1] + Bc * Hc,
      h0pp, h1pp, c0d, c1d, attb);

  // ---- decoder steps ----
  int p0 = 0, p1 = 0;
  for (int t = 0; t < Tc - 1; ++t) {
    build_rnn<<<dim3((Bc * (Ec + DHc) + 255) / 256), 256, 0, stream>>>(emby, attb, rnn, t);
    dec_cell<<<dim3(DHc / 16), 32, 0, stream>>>(
        rnn, Ec + DHc, bWih0, h0pp + (size_t)p0 * Bc * DHc, bWhh0,
        dc0.bih, dc0.bhh, c0d, h0pp + (size_t)(1 - p0) * Bc * DHc,
        nullptr, nullptr, 0);
    p0 ^= 1;
    dec_cell<<<dim3(DHc / 16), 32, 0, stream>>>(
        h0pp + (size_t)p0 * Bc * DHc, DHc, bWih1, h1pp + (size_t)p1 * Bc * DHc, bWhh1,
        dc1.bih, dc1.bhh, c1d, h1pp + (size_t)(1 - p1) * Bc * DHc,
        h1f, ccb + DHc, 2 * DHc);
    p1 ^= 1;
    __bf16* h1b = h1pp + (size_t)p1 * Bc * DHc;
    // tanh(h1 @ W_p^T + b_p) -> bf16
    gemm_wmma<1, 1><<<dim3(1, DHc / 64), 128, 0, stream>>>(
        h1b, DHc, bWp, DHc, nullptr, tpb, DHc, b_p, DHc);
    pt_kernel<<<dim3(Bc), 32, 0, stream>>>(tpb, bvp, b_vp, p_t);
    attn_ctx<<<dim3(Bc), 256, 0, stream>>>(h1f, eproj, ys1, p_t, ccb);
    // att = tanh(concat(ctx,h1) @ W_c^T + b_c) -> bf16
    gemm_wmma<1, 1><<<dim3(1, DHc / 64), 128, 0, stream>>>(
        ccb, 2 * DHc, bWc, 2 * DHc, nullptr, attb, DHc, b_c, 2 * DHc);
    // pred = att @ W_out^T + b_out -> d_out[b, t, :]
    gemm_wmma<0, 0><<<dim3(1, VTGT / 64), 128, 0, stream>>>(
        attb, DHc, bWo, DHc, out + (size_t)t * VTGT, nullptr, (Tc - 1) * VTGT, b_out, DHc);
  }
}
